// Discriminator_54013508714862
// MI455X (gfx1250) — matrix operand order repaired and checked by reading, not compiled, not run on hardware
//
#include <hip/hip_runtime.h>
#include <math.h>

#define NN 4096
#define KB_STEPS 128            // 4096 / 32
#define TILE_ELEMS 512          // 32 lanes * 16 bf16 per (tile, k-step)

typedef __attribute__((ext_vector_type(16))) __bf16 v16bf;
typedef __attribute__((ext_vector_type(8)))  float  v8f;

// ---------------------------------------------------------------------------
// Utility kernels
// ---------------------------------------------------------------------------
__global__ void zero_f32(float* __restrict__ p, size_t n) {
    size_t i = (size_t)blockIdx.x * blockDim.x + threadIdx.x;
    if (i < n) p[i] = 0.0f;
}

__global__ void deg_kernel(const int* __restrict__ dst, float* __restrict__ deg, int E) {
    int e = blockIdx.x * blockDim.x + threadIdx.x;
    if (e < E) atomicAdd(&deg[dst[e]], 1.0f);
}

__global__ void dinv_kernel(const float* __restrict__ deg, float* __restrict__ dinv) {
    int i = blockIdx.x * blockDim.x + threadIdx.x;
    if (i < NN) dinv[i] = rsqrtf(deg[i] + 1.0f);   // +1 self loop; always > 0
}

// ---------------------------------------------------------------------------
// Pack fp32 matrices into bf16 WMMA fragment layout (CDNA5 ISA 7.12.2).
// Packed element index: (((tile*128 + kb)*32 + lane)*16 + e)
//   A (16x32):  value = X[(tile*16 + (lane&15)) , kb*32 + 16*(e>>3) + 8*(lane>>4) + (e&7)]
//   B (32x16):  value = W[(kb*32 + 16*(e>>3) + 8*(lane>>4) + (e&7)) , tile*16 + (lane&15)]
// ---------------------------------------------------------------------------
__global__ void pack_a_bf16(const float* __restrict__ X, __bf16* __restrict__ Xp) {
    unsigned t = blockIdx.x * blockDim.x + threadIdx.x;   // < NN*NN
    int e    = t & 15;
    int lane = (t >> 4) & 31;
    int kb   = (t >> 9) & 127;
    int rt   = t >> 16;
    int row  = rt * 16 + (lane & 15);
    int k    = kb * 32 + 16 * (e >> 3) + 8 * (lane >> 4) + (e & 7);
    Xp[t] = (__bf16)X[(size_t)row * NN + k];
}

__global__ void pack_b_bf16(const float* __restrict__ W, __bf16* __restrict__ Wp) {
    unsigned t = blockIdx.x * blockDim.x + threadIdx.x;   // < NN*NN
    int e    = t & 15;
    int lane = (t >> 4) & 31;
    int kb   = (t >> 9) & 127;
    int ct   = t >> 16;
    int col  = ct * 16 + (lane & 15);
    int k    = kb * 32 + 16 * (e >> 3) + 8 * (lane >> 4) + (e & 7);
    Wp[t] = (__bf16)W[(size_t)k * NN + col];
}

// ---------------------------------------------------------------------------
// GEMM1: H = X @ W1 from pre-packed bf16 fragments.
// Each wave computes a 16x64 output strip: 4 accumulators sharing one A frag,
// 4 v_wmma_f32_16x16x32_bf16 per k-step. All fragment loads are contiguous
// 32B per lane (pairs of global_load_b128).
// ---------------------------------------------------------------------------
__global__ __launch_bounds__(256) void gemm1_wmma(const __bf16* __restrict__ Xp,
                                                  const __bf16* __restrict__ Wp,
                                                  float* __restrict__ H) {
    const int lane = threadIdx.x & 31;
    const int wid  = blockIdx.x * 8 + (threadIdx.x >> 5);  // global wave id
    const int rt   = wid >> 6;       // row tile     0..255
    const int cg   = wid & 63;       // column group 0..63 (4 tiles each)
    const int mn   = lane & 15;
    const int kh   = lane >> 4;

    const __bf16* ap  = Xp + (size_t)rt * KB_STEPS * TILE_ELEMS + lane * 16;
    const __bf16* bp0 = Wp + (size_t)(cg * 4 + 0) * KB_STEPS * TILE_ELEMS + lane * 16;
    const __bf16* bp1 = Wp + (size_t)(cg * 4 + 1) * KB_STEPS * TILE_ELEMS + lane * 16;
    const __bf16* bp2 = Wp + (size_t)(cg * 4 + 2) * KB_STEPS * TILE_ELEMS + lane * 16;
    const __bf16* bp3 = Wp + (size_t)(cg * 4 + 3) * KB_STEPS * TILE_ELEMS + lane * 16;

    v8f acc0 = {}, acc1 = {}, acc2 = {}, acc3 = {};

    for (int kb = 0; kb < KB_STEPS; ++kb) {
        size_t o = (size_t)kb * TILE_ELEMS;
        v16bf a  = *(const v16bf*)(ap  + o);
        v16bf b0 = *(const v16bf*)(bp0 + o);
        v16bf b1 = *(const v16bf*)(bp1 + o);
        v16bf b2 = *(const v16bf*)(bp2 + o);
        v16bf b3 = *(const v16bf*)(bp3 + o);

        acc0 = __builtin_amdgcn_wmma_f32_16x16x32_bf16(false, a, false, b0,
                                                       (short)0, acc0, false, false);
        acc1 = __builtin_amdgcn_wmma_f32_16x16x32_bf16(false, a, false, b1,
                                                       (short)0, acc1, false, false);
        acc2 = __builtin_amdgcn_wmma_f32_16x16x32_bf16(false, a, false, b2,
                                                       (short)0, acc2, false, false);
        acc3 = __builtin_amdgcn_wmma_f32_16x16x32_bf16(false, a, false, b3,
                                                       (short)0, acc3, false, false);

        if (kb + 1 < KB_STEPS) {   // prefetch next k-step (global_prefetch_b8)
            size_t on = o + TILE_ELEMS;
            __builtin_prefetch(ap  + on, 0, 3);
            __builtin_prefetch(bp0 + on, 0, 3);
            __builtin_prefetch(bp2 + on, 0, 3);
        }
    }

    // D store: VGPR r -> row (rt*16 + r + 8*kh), col (tile*16 + mn)
    float* hb = H + (size_t)(rt * 16 + 8 * kh) * NN + cg * 64 + mn;
    #pragma unroll
    for (int r = 0; r < 8; ++r) {
        hb[(size_t)r * NN +  0] = acc0[r];
        hb[(size_t)r * NN + 16] = acc1[r];
        hb[(size_t)r * NN + 32] = acc2[r];
        hb[(size_t)r * NN + 48] = acc3[r];
    }
}

// ---------------------------------------------------------------------------
// Edge aggregation, layer 1: agg1[dst,:] += H[src,:] * dinv[src]*dinv[dst]
// One block per edge; atomics resolve in L2 (all buffers fit in 192MB L2).
// ---------------------------------------------------------------------------
__global__ void edge_agg1(const int* __restrict__ src, const int* __restrict__ dst,
                          const float* __restrict__ dinv, const float* __restrict__ H,
                          float* __restrict__ agg1) {
    int e = blockIdx.x;
    int s = src[e], d = dst[e];
    float norm = dinv[s] * dinv[d];
    const float* hs = H + (size_t)s * NN;
    float* od = agg1 + (size_t)d * NN;
    for (int f = threadIdx.x; f < NN; f += blockDim.x)
        atomicAdd(&od[f], hs[f] * norm);
}

// Fused self-loop + bias + sigmoid: x1 = sigmoid(agg1 + H*dinv^2 + b1), in place.
__global__ void act1_kernel(float* __restrict__ agg1, const float* __restrict__ H,
                            const float* __restrict__ dinv, const float* __restrict__ b1) {
    size_t i = (size_t)blockIdx.x * blockDim.x + threadIdx.x;
    int row = (int)(i >> 12);            // /4096
    int col = (int)(i & (NN - 1));
    float di = dinv[row];
    float v = agg1[i] + H[i] * di * di + b1[col];
    agg1[i] = 1.0f / (1.0f + __expf(-v));
}

// ---------------------------------------------------------------------------
// Layer 2 GEMV: z[row] = dot(x1[row,:], W2)
// ---------------------------------------------------------------------------
__global__ __launch_bounds__(256) void gemv2_kernel(const float* __restrict__ x1,
                                                    const float* __restrict__ W2,
                                                    float* __restrict__ z) {
    __shared__ float red[256];
    int row = blockIdx.x;
    int tid = threadIdx.x;
    const float* xr = x1 + (size_t)row * NN;
    float s = 0.0f;
    for (int k = tid; k < NN; k += 256) s += xr[k] * W2[k];
    red[tid] = s;
    __syncthreads();
    for (int off = 128; off > 0; off >>= 1) {
        if (tid < off) red[tid] += red[tid + off];
        __syncthreads();
    }
    if (tid == 0) z[row] = red[0];
}

__global__ void edge_agg2(const int* __restrict__ src, const int* __restrict__ dst,
                          const float* __restrict__ dinv, const float* __restrict__ z,
                          float* __restrict__ agg2, int E) {
    int e = blockIdx.x * blockDim.x + threadIdx.x;
    if (e < E) {
        int s = src[e], d = dst[e];
        atomicAdd(&agg2[d], z[s] * dinv[s] * dinv[d]);
    }
}

__global__ void act2_kernel(const float* __restrict__ agg2, const float* __restrict__ z,
                            const float* __restrict__ dinv, const float* __restrict__ b2,
                            float* __restrict__ out) {
    int i = blockIdx.x * blockDim.x + threadIdx.x;
    if (i < NN) {
        float di = dinv[i];
        float v = agg2[i] + z[i] * di * di + b2[0];
        out[i] = 1.0f / (1.0f + __expf(-v));
    }
}

// ---------------------------------------------------------------------------
extern "C" void kernel_launch(void* const* d_in, const int* in_sizes, int n_in,
                              void* d_out, int out_size, void* d_ws, size_t ws_size,
                              hipStream_t stream) {
    const float* x  = (const float*)d_in[0];
    const int*   ei = (const int*)  d_in[1];
    const float* W1 = (const float*)d_in[2];
    const float* b1 = (const float*)d_in[3];
    const float* W2 = (const float*)d_in[4];
    const float* b2 = (const float*)d_in[5];
    const int E = in_sizes[1] / 2;
    const int* src = ei;
    const int* dst = ei + E;

    // Workspace layout: H (64MB) | agg1 (64MB, aliased by Wp+Xp during GEMM) |
    //                   deg | dinv | z | agg2  (4 x 16KB)
    float*  H    = (float*)d_ws;
    float*  agg1 = H + (size_t)NN * NN;
    __bf16* Wp   = (__bf16*)agg1;                       // 32MB, GEMM phase only
    __bf16* Xp   = Wp + (size_t)NN * NN;                // 32MB, GEMM phase only
    float*  deg  = agg1 + (size_t)NN * NN;
    float*  dinv = deg  + NN;
    float*  z    = dinv + NN;
    float*  agg2 = z    + NN;

    const unsigned NTOT = (unsigned)NN * NN;            // 16,777,216

    // Zero small tail region (deg | dinv | z | agg2) before degree pass.
    zero_f32<<<(4 * NN) / 256, 256, 0, stream>>>(deg, 4 * (size_t)NN);

    deg_kernel<<<(E + 255) / 256, 256, 0, stream>>>(dst, deg, E);
    dinv_kernel<<<NN / 256, 256, 0, stream>>>(deg, dinv);

    // Pack to bf16 fragment layout, then WMMA GEMM.
    pack_a_bf16<<<NTOT / 256, 256, 0, stream>>>(x,  Xp);
    pack_b_bf16<<<NTOT / 256, 256, 0, stream>>>(W1, Wp);
    gemm1_wmma<<<2048, 256, 0, stream>>>(Xp, Wp, H);

    // Pack buffers dead now; reuse the region as agg1 (zero it first).
    zero_f32<<<NTOT / 256, 256, 0, stream>>>(agg1, (size_t)NTOT);

    edge_agg1<<<E, 256, 0, stream>>>(src, dst, dinv, H, agg1);
    act1_kernel<<<NTOT / 256, 256, 0, stream>>>(agg1, H, dinv, b1);

    gemv2_kernel<<<NN, 256, 0, stream>>>(agg1, W2, z);
    edge_agg2<<<(E + 255) / 256, 256, 0, stream>>>(src, dst, dinv, z, agg2, E);
    act2_kernel<<<NN / 256, 256, 0, stream>>>(agg2, z, dinv, b2, (float*)d_out);
}
